// FrequencyMamba_47991964565852
// MI455X (gfx1250) — matrix operand order repaired and checked by reading, not compiled, not run on hardware
//
#include <hip/hip_runtime.h>
#include <math.h>

typedef _Float16 h16;
typedef __attribute__((ext_vector_type(16))) _Float16 v16h;
typedef __attribute__((ext_vector_type(8)))  _Float16 v8h;
typedef __attribute__((ext_vector_type(8)))  float    v8f;

namespace fm {
constexpr int nVAR = 3;           // x, coarse, fine
constexpr int nB   = 32;
constexpr int nWIN = 512;
constexpr int nCH  = 64;
constexpr int nDM  = 128;         // D_MODEL (== 2*nCH)
constexpr int nDI  = 256;         // D_INNER
constexpr int nDS  = 16;          // D_STATE
constexpr int nDTR = 8;           // DT_RANK
constexpr int nKC  = 4;           // K_CONV
constexpr int nF   = 257;         // WIN/2+1
constexpr int nFPAD = 272;        // 17*16 (M padding for WMMA tiles)
constexpr int nKIR = 576;         // irfft K: [0,257) real, [288,545) imag, rest zero-weighted
constexpr float AUG = 0.3f;
constexpr float INV_TEMP = 2.0f;  // 1/0.5
constexpr float PI2 = 6.28318530717958647692f;
}
using namespace fm;

__device__ __forceinline__ float siluf(float x) { return x / (1.0f + __expf(-x)); }

// ---------------------------------------------------------------------------
// Wave-level 16x16 tile GEMM: C += A(MxK, row-major f16) * B(KxN) where B is
// supplied transposed (Bt: NxK row-major f16). K must be a multiple of 32.
// Fragment layouts per CDNA5 ISA 7.12.2 (wave32):
//  A lane (m = lane&15, hi = lane>>4): halves [0..7] = K 8*hi..8*hi+7,
//                                      halves [8..15] = K 16+8*hi..16+8*hi+7
//  B lane (n = lane&15, hi = lane>>4): same K mapping as A
//  D lane: row = v + 8*hi, col = lane&15
// ---------------------------------------------------------------------------
__device__ __forceinline__ v8f wmma_tile(const h16* __restrict__ A, int lda,
                                         const h16* __restrict__ Bt, int ldb,
                                         int row0, int col0, int K, v8f acc) {
  const int lane = threadIdx.x & 31;
  const int rr = lane & 15;
  const int hi = lane >> 4;
  const h16* ap = A + (size_t)(row0 + rr) * lda + hi * 8;
  const h16* bp = Bt + (size_t)(col0 + rr) * ldb + hi * 8;
  for (int k = 0; k < K; k += 32) {
    v8h a0 = *(const v8h*)(ap + k);
    v8h a1 = *(const v8h*)(ap + k + 16);
    v8h b0 = *(const v8h*)(bp + k);
    v8h b1 = *(const v8h*)(bp + k + 16);
    v16h va, vb;
#pragma unroll
    for (int i = 0; i < 8; ++i) {
      va[i] = a0[i]; va[i + 8] = a1[i];
      vb[i] = b0[i]; vb[i + 8] = b1[i];
    }
    acc = __builtin_amdgcn_wmma_f32_16x16x32_f16(false, va, false, vb,
                                                 (short)0, acc, false, false);
  }
  return acc;
}

// ------------------------------- prep kernels ------------------------------

// DFT tables: Wc[f][t] = cos(2*pi*f*t/512), Ws[f][t] = -sin(...)  (f<257)
__global__ void k_prep_dft(h16* Wc, h16* Ws) {
  int idx = blockIdx.x * blockDim.x + threadIdx.x;
  if (idx >= nFPAD * nWIN) return;
  int f = idx / nWIN, t = idx % nWIN;
  float c = 0.f, s = 0.f;
  if (f < nF) {
    int ph = (f * t) & (nWIN - 1);
    float th = PI2 * (float)ph / (float)nWIN;
    c = cosf(th); s = -sinf(th);
  }
  Wc[idx] = (h16)c; Ws[idx] = (h16)s;
}

// irfft matrix A2 (WIN x KIR): out[t] = sum_k A2[t][k]*Bir[k]
// k in [0,288): real bin f=k, weight (f==0||f==256)?1:2, cos
// k in [288,576): imag bin f=k-288, weight (0<f<256)?-2:0, sin
// (numpy/jax irfft ignores Im[0], Im[N/2])
__global__ void k_prep_irfft(h16* A2) {
  int idx = blockIdx.x * blockDim.x + threadIdx.x;
  if (idx >= nWIN * nKIR) return;
  int t = idx / nKIR, k = idx % nKIR;
  float v = 0.f;
  if (k < 288) {
    int f = k;
    if (f < nF) {
      float wgt = (f == 0 || f == nF - 1) ? 1.f : 2.f;
      int ph = (f * t) & (nWIN - 1);
      v = wgt * cosf(PI2 * (float)ph / (float)nWIN) / (float)nWIN;
    }
  } else {
    int f = k - 288;
    if (f >= 1 && f < nF - 1) {
      int ph = (f * t) & (nWIN - 1);
      v = -2.f * sinf(PI2 * (float)ph / (float)nWIN) / (float)nWIN;
    }
  }
  A2[idx] = (h16)v;
}

// dst[n][k] = src[k][n] (f32 -> f16), n < Npad rows, zero-fill n >= N
__global__ void k_transpose(const float* __restrict__ src, h16* __restrict__ dst,
                            int K, int N, int Npad) {
  int idx = blockIdx.x * blockDim.x + threadIdx.x;
  if (idx >= Npad * K) return;
  int k = idx % K;
  int n = idx / K;
  dst[idx] = (n < N) ? (h16)src[(size_t)k * N + n] : (h16)0.f;
}

// Build the 3 augmented signals, transposed+f16: xT[v][b][c][t]
__global__ void k_aug(const float* __restrict__ x, const float* __restrict__ jit,
                      const float* __restrict__ um, h16* __restrict__ xT) {
  int idx = blockIdx.x * blockDim.x + threadIdx.x;
  int total = nVAR * nB * nWIN * nCH;
  if (idx >= total) return;
  int c = idx % nCH;
  int r = idx / nCH;
  int t = r % nWIN;
  int vb = r / nWIN;
  int v = vb / nB, b = vb % nB;
  size_t xi = ((size_t)b * nWIN + t) * nCH + c;
  float val = x[xi];
  if (v == 1) val = (um[(size_t)b * nWIN + t] > AUG) ? val : 0.f;
  else if (v == 2) val = val + AUG * jit[xi];
  xT[((size_t)vb * nCH + c) * nWIN + t] = (h16)val;
}

__global__ void k_zero16(h16* p, int n) {
  int idx = blockIdx.x * blockDim.x + threadIdx.x;
  if (idx < n) p[idx] = (h16)0.f;
}

// ------------------------------- GEMM kernels ------------------------------

// DFT: per (v,b): Re/Im (F x CH) = Wcos/Wsin (F x WIN) @ x^T^T; Bt = xT[v][b]
__global__ void k_dft(const h16* __restrict__ Wc, const h16* __restrict__ Ws,
                      const h16* __restrict__ xT, h16* __restrict__ cat16,
                      float* __restrict__ Re32, float* __restrict__ Im32) {
  int wave = threadIdx.x >> 5;
  int tile = blockIdx.x * 8 + wave;
  const int ntn = nCH / 16;                       // 4
  if (tile >= (nFPAD / 16) * ntn) return;
  int vb = blockIdx.y;
  int v = vb / nB, b = vb % nB;
  int tm = tile / ntn, tn = tile % ntn;
  const h16* Bt = xT + (size_t)vb * nCH * nWIN;
  v8f ar = {}, ai = {};
  ar = wmma_tile(Wc, nWIN, Bt, nWIN, tm * 16, tn * 16, nWIN, ar);
  ai = wmma_tile(Ws, nWIN, Bt, nWIN, tm * 16, tn * 16, nWIN, ai);
  int lane = threadIdx.x & 31, c = lane & 15, hi = lane >> 4;
#pragma unroll
  for (int e = 0; e < 8; ++e) {
    int f = tm * 16 + e + hi * 8;
    int ch = tn * 16 + c;
    if (f < nF) {
      size_t cb = ((size_t)vb * nFPAD + f) * nDM;
      cat16[cb + ch] = (h16)ar[e];
      cat16[cb + nCH + ch] = (h16)ai[e];
      if (v == 0) {
        size_t ri = ((size_t)b * nF + f) * nCH + ch;
        Re32[ri] = ar[e];
        Im32[ri] = ai[e];
      }
    }
  }
}

// emb = cat @ fre_w + fre_b   (M=F, N=128, K=128)
__global__ void k_emb(const h16* __restrict__ cat16, const h16* __restrict__ freT,
                      const float* __restrict__ fre_b, h16* __restrict__ emb16) {
  int wave = threadIdx.x >> 5;
  int tile = blockIdx.x * 8 + wave;
  const int ntn = nDM / 16;                       // 8
  if (tile >= (nFPAD / 16) * ntn) return;
  int vb = blockIdx.y;
  int tm = tile / ntn, tn = tile % ntn;
  const h16* A = cat16 + (size_t)vb * nFPAD * nDM;
  v8f acc = {};
  acc = wmma_tile(A, nDM, freT, nDM, tm * 16, tn * 16, nDM, acc);
  int lane = threadIdx.x & 31, c = lane & 15, hi = lane >> 4;
#pragma unroll
  for (int e = 0; e < 8; ++e) {
    int f = tm * 16 + e + hi * 8, n = tn * 16 + c;
    if (f < nF)
      emb16[((size_t)vb * nFPAD + f) * nDM + n] = (h16)(acc[e] + fre_b[n]);
  }
}

// xz = emb @ in_proj_w  (N=512): n<256 -> x_in f32, n>=256 -> z f16
__global__ void k_inproj(const h16* __restrict__ emb16, const h16* __restrict__ inpT,
                         float* __restrict__ xin, h16* __restrict__ z16) {
  int wave = threadIdx.x >> 5;
  int tile = blockIdx.x * 8 + wave;
  const int ntn = (2 * nDI) / 16;                 // 32
  if (tile >= (nFPAD / 16) * ntn) return;
  int vb = blockIdx.y;
  int tm = tile / ntn, tn = tile % ntn;
  const h16* A = emb16 + (size_t)vb * nFPAD * nDM;
  v8f acc = {};
  acc = wmma_tile(A, nDM, inpT, nDM, tm * 16, tn * 16, nDM, acc);
  int lane = threadIdx.x & 31, c = lane & 15, hi = lane >> 4;
#pragma unroll
  for (int e = 0; e < 8; ++e) {
    int f = tm * 16 + e + hi * 8, n = tn * 16 + c;
    if (f < nF) {
      if (n < nDI) xin[((size_t)vb * nF + f) * nDI + n] = acc[e];
      else z16[((size_t)vb * nF + f) * nDI + (n - nDI)] = (h16)acc[e];
    }
  }
}

// causal depthwise conv (K=4) + bias + silu -> xc (f16, padded rows)
__global__ void k_conv(const float* __restrict__ xin, const float* __restrict__ conv_w,
                       const float* __restrict__ conv_b, h16* __restrict__ xc16) {
  int idx = blockIdx.x * blockDim.x + threadIdx.x;
  int total = nVAR * nB * nF * nDI;
  if (idx >= total) return;
  int d = idx % nDI;
  int r = idx / nDI;
  int f = r % nF;
  int vb = r / nF;
  float s = conv_b[d];
#pragma unroll
  for (int j = 0; j < nKC; ++j) {
    int t = f - (nKC - 1) + j;
    if (t >= 0) s += xin[((size_t)vb * nF + t) * nDI + d] * conv_w[d * nKC + j];
  }
  xc16[((size_t)vb * nFPAD + f) * nDI + d] = (h16)siluf(s);
}

// x_dbl = xc @ xproj_w (N=40 padded to 48, K=256)
__global__ void k_xproj(const h16* __restrict__ xc16, const h16* __restrict__ xpT,
                        float* __restrict__ xdbl) {
  int wave = threadIdx.x >> 5;
  int tile = blockIdx.x * 8 + wave;
  const int ntn = 3;                              // 48/16
  if (tile >= (nFPAD / 16) * ntn) return;
  int vb = blockIdx.y;
  int tm = tile / ntn, tn = tile % ntn;
  const h16* A = xc16 + (size_t)vb * nFPAD * nDI;
  v8f acc = {};
  acc = wmma_tile(A, nDI, xpT, nDI, tm * 16, tn * 16, nDI, acc);
  int lane = threadIdx.x & 31, c = lane & 15, hi = lane >> 4;
#pragma unroll
  for (int e = 0; e < 8; ++e) {
    int f = tm * 16 + e + hi * 8, n = tn * 16 + c;
    if (f < nF) xdbl[((size_t)vb * nF + f) * 48 + n] = acc[e];
  }
}

// Selective scan: one block per (v,b), one thread per d; state in registers.
// Fuses dt-projection (K=8), softplus, dA/dBu, C-contraction, D-skip, silu(z) gate.
__global__ void k_scan(const float* __restrict__ xdbl, const h16* __restrict__ xc16,
                       const h16* __restrict__ z16, const float* __restrict__ dtw,
                       const float* __restrict__ dtb, const float* __restrict__ A_log,
                       const float* __restrict__ Dp, h16* __restrict__ so16) {
  int vb = blockIdx.x;
  int d = threadIdx.x;
  float w[nDTR];
#pragma unroll
  for (int r = 0; r < nDTR; ++r) w[r] = dtw[r * nDI + d];
  float bias = dtb[d];
  float Dv = Dp[d];
  float Aa[nDS], h[nDS];
#pragma unroll
  for (int s = 0; s < nDS; ++s) { Aa[s] = -__expf(A_log[d * nDS + s]); h[s] = 0.f; }
  for (int f = 0; f < nF; ++f) {
    const float* xd = xdbl + ((size_t)vb * nF + f) * 48;
    float dtv = bias;
#pragma unroll
    for (int r = 0; r < nDTR; ++r) dtv += xd[r] * w[r];
    dtv = (dtv > 20.f) ? dtv : log1pf(__expf(dtv));
    float xcv = (float)xc16[((size_t)vb * nFPAD + f) * nDI + d];
    float y = 0.f;
#pragma unroll
    for (int s = 0; s < nDS; ++s) {
      h[s] = __expf(dtv * Aa[s]) * h[s] + dtv * xd[8 + s] * xcv;
      y += h[s] * xd[24 + s];
    }
    float zv = (float)z16[((size_t)vb * nF + f) * nDI + d];
    so16[((size_t)vb * nFPAD + f) * nDI + d] = (h16)((y + xcv * Dv) * siluf(zv));
  }
}

// enc = gated @ out_proj_w (N=128, K=256)
__global__ void k_outproj(const h16* __restrict__ so16, const h16* __restrict__ opT,
                          h16* __restrict__ enc16) {
  int wave = threadIdx.x >> 5;
  int tile = blockIdx.x * 8 + wave;
  const int ntn = nDM / 16;                       // 8
  if (tile >= (nFPAD / 16) * ntn) return;
  int vb = blockIdx.y;
  int tm = tile / ntn, tn = tile % ntn;
  const h16* A = so16 + (size_t)vb * nFPAD * nDI;
  v8f acc = {};
  acc = wmma_tile(A, nDI, opT, nDI, tm * 16, tn * 16, nDI, acc);
  int lane = threadIdx.x & 31, c = lane & 15, hi = lane >> 4;
#pragma unroll
  for (int e = 0; e < 8; ++e) {
    int f = tm * 16 + e + hi * 8, n = tn * 16 + c;
    if (f < nF) enc16[((size_t)vb * nFPAD + f) * nDM + n] = (h16)acc[e];
  }
}

// heads (variant 0 only): real/imag = enc @ getr/geti + b; also writes Bt for irfft
__global__ void k_heads(const h16* __restrict__ enc16, const h16* __restrict__ grT,
                        const float* __restrict__ grb, const h16* __restrict__ giT,
                        const float* __restrict__ gib, float* __restrict__ real32,
                        float* __restrict__ imag32, h16* __restrict__ BirT) {
  int wave = threadIdx.x >> 5;
  int tile = blockIdx.x * 8 + wave;
  const int ntn = nCH / 16;                       // 4
  if (tile >= (nFPAD / 16) * ntn) return;
  int b = blockIdx.y;
  int tm = tile / ntn, tn = tile % ntn;
  const h16* A = enc16 + (size_t)b * nFPAD * nDM; // variant 0 slice
  v8f ar = {}, ai = {};
  ar = wmma_tile(A, nDM, grT, nDM, tm * 16, tn * 16, nDM, ar);
  ai = wmma_tile(A, nDM, giT, nDM, tm * 16, tn * 16, nDM, ai);
  int lane = threadIdx.x & 31, c = lane & 15, hi = lane >> 4;
#pragma unroll
  for (int e = 0; e < 8; ++e) {
    int f = tm * 16 + e + hi * 8, ch = tn * 16 + c;
    if (f < nF) {
      float vr = ar[e] + grb[ch];
      float vi = ai[e] + gib[ch];
      size_t ri = ((size_t)b * nF + f) * nCH + ch;
      real32[ri] = vr; imag32[ri] = vi;
      size_t bi = ((size_t)b * nCH + ch) * nKIR;
      BirT[bi + f] = (h16)vr;
      BirT[bi + 288 + f] = (h16)vi;
    }
  }
}

// output = A2 (512 x 576) @ Bir (576 x 64) per batch -> d_out
__global__ void k_irfft(const h16* __restrict__ A2, const h16* __restrict__ BirT,
                        float* __restrict__ out) {
  int wave = threadIdx.x >> 5;
  int tile = blockIdx.x * 8 + wave;
  const int ntn = nCH / 16;                       // 4
  if (tile >= (nWIN / 16) * ntn) return;
  int b = blockIdx.y;
  int tm = tile / ntn, tn = tile % ntn;
  v8f acc = {};
  acc = wmma_tile(A2, nKIR, BirT + (size_t)b * nCH * nKIR, nKIR,
                  tm * 16, tn * 16, nKIR, acc);
  int lane = threadIdx.x & 31, c = lane & 15, hi = lane >> 4;
#pragma unroll
  for (int e = 0; e < 8; ++e) {
    int t = tm * 16 + e + hi * 8, ch = tn * 16 + c;
    out[((size_t)b * nWIN + t) * nCH + ch] = acc[e];
  }
}

// ------------------------------ loss kernels -------------------------------

__global__ void k_norm2(const h16* __restrict__ enc16, float* __restrict__ norm2) {
  __shared__ float red[256];
  int vb = blockIdx.x;
  const h16* p = enc16 + (size_t)vb * nFPAD * nDM;
  float s = 0.f;
  for (int n = threadIdx.x; n < nF * nDM; n += 256) {
    int f = n >> 7, c = n & 127;
    float v = (float)p[(size_t)f * nDM + c];
    s += v * v;
  }
  red[threadIdx.x] = s; __syncthreads();
  for (int off = 128; off > 0; off >>= 1) {
    if ((int)threadIdx.x < off) red[threadIdx.x] += red[threadIdx.x + off];
    __syncthreads();
  }
  if (threadIdx.x == 0) norm2[vb] = red[0];
}

// dots[p][i][j] = <enc0_i, enc(p+1)_j> over flattened (F, 128)
__global__ void k_dots(const h16* __restrict__ enc16, float* __restrict__ dots) {
  __shared__ float red[256];
  int p = blockIdx.y;
  int i = blockIdx.x / nB, j = blockIdx.x % nB;
  const h16* pa = enc16 + (size_t)i * nFPAD * nDM;
  const h16* pb = enc16 + ((size_t)(p + 1) * nB + j) * nFPAD * nDM;
  float s = 0.f;
  for (int n = threadIdx.x; n < nF * nDM; n += 256) {
    int f = n >> 7, c = n & 127;
    s += (float)pa[(size_t)f * nDM + c] * (float)pb[(size_t)f * nDM + c];
  }
  red[threadIdx.x] = s; __syncthreads();
  for (int off = 128; off > 0; off >>= 1) {
    if ((int)threadIdx.x < off) red[threadIdx.x] += red[threadIdx.x + off];
    __syncthreads();
  }
  if (threadIdx.x == 0) dots[((size_t)p * nB + i) * nB + j] = red[0];
}

__global__ void k_loss(const float* __restrict__ dots, const float* __restrict__ norm2,
                       float* __restrict__ out_loss) {
  __shared__ float red[32];
  int i = threadIdx.x;
  float n1 = sqrtf(norm2[i]);
  float acc = 0.f;
  for (int p = 0; p < 2; ++p) {
    float ssum = 0.f, pos = 1.f;
    for (int j = 0; j < nB; ++j) {
      float sim = dots[((size_t)p * nB + i) * nB + j] /
                  (n1 * sqrtf(norm2[(p + 1) * nB + j]));
      float e = __expf(sim * INV_TEMP);
      ssum += e;
      if (j == i) pos = e;
    }
    acc += -logf(pos / (ssum - pos));
  }
  red[i] = acc / (float)nB;
  __syncthreads();
  if (i == 0) {
    float t = 0.f;
    for (int j = 0; j < nB; ++j) t += red[j];
    *out_loss = t;
  }
}

__global__ void k_recon(const float* __restrict__ Re, const float* __restrict__ Im,
                        const float* __restrict__ real32, const float* __restrict__ imag32,
                        float* __restrict__ partial) {
  __shared__ float red[256];
  int total = nB * nF * nCH;
  float s = 0.f;
  for (int idx = blockIdx.x * 256 + threadIdx.x; idx < total; idx += gridDim.x * 256) {
    float dr = Re[idx] - real32[idx];
    float di = Im[idx] - imag32[idx];
    s += dr * dr + di * di;
  }
  red[threadIdx.x] = s; __syncthreads();
  for (int off = 128; off > 0; off >>= 1) {
    if ((int)threadIdx.x < off) red[threadIdx.x] += red[threadIdx.x + off];
    __syncthreads();
  }
  if (threadIdx.x == 0) partial[blockIdx.x] = red[0];
}

__global__ void k_recon_final(const float* __restrict__ partial, int np,
                              float* __restrict__ out) {
  __shared__ float red[256];
  float s = 0.f;
  for (int i = threadIdx.x; i < np; i += 256) s += partial[i];
  red[threadIdx.x] = s; __syncthreads();
  for (int off = 128; off > 0; off >>= 1) {
    if ((int)threadIdx.x < off) red[threadIdx.x] += red[threadIdx.x + off];
    __syncthreads();
  }
  if (threadIdx.x == 0) *out = red[0] / (float)(nB * nF * nCH);
}

// --------------------------------- launch ----------------------------------

extern "C" void kernel_launch(void* const* d_in, const int* in_sizes, int n_in,
                              void* d_out, int out_size, void* d_ws, size_t ws_size,
                              hipStream_t stream) {
  const float* x         = (const float*)d_in[0];
  const float* jitter    = (const float*)d_in[1];
  const float* u_mask    = (const float*)d_in[2];
  const float* fre_w     = (const float*)d_in[3];
  const float* fre_b     = (const float*)d_in[4];
  const float* in_proj_w = (const float*)d_in[5];
  const float* conv_w    = (const float*)d_in[6];
  const float* conv_b    = (const float*)d_in[7];
  const float* xproj_w   = (const float*)d_in[8];
  const float* dtproj_w  = (const float*)d_in[9];
  const float* dtproj_b  = (const float*)d_in[10];
  const float* A_log     = (const float*)d_in[11];
  const float* D_param   = (const float*)d_in[12];
  const float* out_proj_w= (const float*)d_in[13];
  const float* getr_w    = (const float*)d_in[14];
  const float* getr_b    = (const float*)d_in[15];
  const float* geti_w    = (const float*)d_in[16];
  const float* geti_b    = (const float*)d_in[17];
  (void)in_sizes; (void)n_in; (void)out_size; (void)ws_size;
  float* out = (float*)d_out;

  char* wsb = (char*)d_ws;
  size_t off = 0;
  auto alloc = [&](size_t bytes) -> void* {
    void* p = wsb + off;
    off = (off + bytes + 255) & ~(size_t)255;
    return p;
  };
  h16*   xT    = (h16*)  alloc((size_t)nVAR * nB * nCH * nWIN * 2);
  h16*   Wc    = (h16*)  alloc((size_t)nFPAD * nWIN * 2);
  h16*   Wsn   = (h16*)  alloc((size_t)nFPAD * nWIN * 2);
  h16*   A2    = (h16*)  alloc((size_t)nWIN * nKIR * 2);
  h16*   freT  = (h16*)  alloc((size_t)nDM * nDM * 2);
  h16*   inpT  = (h16*)  alloc((size_t)(2 * nDI) * nDM * 2);
  h16*   xpT   = (h16*)  alloc((size_t)48 * nDI * 2);
  h16*   opT   = (h16*)  alloc((size_t)nDM * nDI * 2);
  h16*   grT   = (h16*)  alloc((size_t)nCH * nDM * 2);
  h16*   giT   = (h16*)  alloc((size_t)nCH * nDM * 2);
  h16*   cat16 = (h16*)  alloc((size_t)nVAR * nB * nFPAD * nDM * 2);
  float* Re32  = (float*)alloc((size_t)nB * nF * nCH * 4);
  float* Im32  = (float*)alloc((size_t)nB * nF * nCH * 4);
  h16*   emb16 = (h16*)  alloc((size_t)nVAR * nB * nFPAD * nDM * 2);
  float* xin   = (float*)alloc((size_t)nVAR * nB * nF * nDI * 4);
  h16*   z16   = (h16*)  alloc((size_t)nVAR * nB * nF * nDI * 2);
  h16*   xc16  = (h16*)  alloc((size_t)nVAR * nB * nFPAD * nDI * 2);
  float* xdbl  = (float*)alloc((size_t)nVAR * nB * nF * 48 * 4);
  h16*   so16  = (h16*)  alloc((size_t)nVAR * nB * nFPAD * nDI * 2);
  h16*   enc16 = (h16*)  alloc((size_t)nVAR * nB * nFPAD * nDM * 2);
  float* real32= (float*)alloc((size_t)nB * nF * nCH * 4);
  float* imag32= (float*)alloc((size_t)nB * nF * nCH * 4);
  h16*   BirT  = (h16*)  alloc((size_t)nB * nCH * nKIR * 2);
  float* dots  = (float*)alloc((size_t)2 * nB * nB * 4);
  float* nrm2  = (float*)alloc((size_t)nVAR * nB * 4);
  float* part  = (float*)alloc((size_t)256 * 4);

  const int T = 256;
  k_prep_dft<<<(nFPAD * nWIN + T - 1) / T, T, 0, stream>>>(Wc, Wsn);
  k_prep_irfft<<<(nWIN * nKIR + T - 1) / T, T, 0, stream>>>(A2);
  k_transpose<<<(nDM * nDM + T - 1) / T, T, 0, stream>>>(fre_w, freT, nDM, nDM, nDM);
  k_transpose<<<(2 * nDI * nDM + T - 1) / T, T, 0, stream>>>(in_proj_w, inpT, nDM, 2 * nDI, 2 * nDI);
  k_transpose<<<(48 * nDI + T - 1) / T, T, 0, stream>>>(xproj_w, xpT, nDI, 40, 48);
  k_transpose<<<(nDM * nDI + T - 1) / T, T, 0, stream>>>(out_proj_w, opT, nDI, nDM, nDM);
  k_transpose<<<(nCH * nDM + T - 1) / T, T, 0, stream>>>(getr_w, grT, nDM, nCH, nCH);
  k_transpose<<<(nCH * nDM + T - 1) / T, T, 0, stream>>>(geti_w, giT, nDM, nCH, nCH);
  k_aug<<<(nVAR * nB * nWIN * nCH + T - 1) / T, T, 0, stream>>>(x, jitter, u_mask, xT);
  k_zero16<<<(nB * nCH * nKIR + T - 1) / T, T, 0, stream>>>(BirT, nB * nCH * nKIR);

  dim3 gdft(9, nVAR * nB);
  k_dft<<<gdft, T, 0, stream>>>(Wc, Wsn, xT, cat16, Re32, Im32);
  dim3 gemb(17, nVAR * nB);
  k_emb<<<gemb, T, 0, stream>>>(cat16, freT, fre_b, emb16);
  dim3 ginp(68, nVAR * nB);
  k_inproj<<<ginp, T, 0, stream>>>(emb16, inpT, xin, z16);
  k_conv<<<(nVAR * nB * nF * nDI + T - 1) / T, T, 0, stream>>>(xin, conv_w, conv_b, xc16);
  dim3 gxp(7, nVAR * nB);
  k_xproj<<<gxp, T, 0, stream>>>(xc16, xpT, xdbl);
  k_scan<<<nVAR * nB, nDI, 0, stream>>>(xdbl, xc16, z16, dtproj_w, dtproj_b, A_log,
                                        D_param, so16);
  dim3 gop(17, nVAR * nB);
  k_outproj<<<gop, T, 0, stream>>>(so16, opT, enc16);

  k_norm2<<<nVAR * nB, T, 0, stream>>>(enc16, nrm2);
  dim3 gdot(nB * nB, 2);
  k_dots<<<gdot, T, 0, stream>>>(enc16, dots);
  k_loss<<<1, 32, 0, stream>>>(dots, nrm2, out + (size_t)nB * nWIN * nCH);

  dim3 ghd(9, nB);
  k_heads<<<ghd, T, 0, stream>>>(enc16, grT, getr_b, giT, geti_b, real32, imag32, BirT);
  dim3 gir(16, nB);
  k_irfft<<<gir, T, 0, stream>>>(A2, BirT, out);

  k_recon<<<256, T, 0, stream>>>(Re32, Im32, real32, imag32, part);
  k_recon_final<<<1, T, 0, stream>>>(part, 256, out + (size_t)nB * nWIN * nCH + 1);
}
